// AttnNet_84593675862295
// MI455X (gfx1250) — matrix operand order repaired and checked by reading, not compiled, not run on hardware
//
#include <hip/hip_runtime.h>
#include <cmath>

typedef __attribute__((ext_vector_type(16))) _Float16 v16h;
typedef __attribute__((ext_vector_type(8)))  _Float16 v8h;
typedef __attribute__((ext_vector_type(4)))  _Float16 v4h;
typedef __attribute__((ext_vector_type(8)))  float    v8f;

#define BATCH 32
#define SEQ   4096
#define HDIM  512
#define MTILE 128
#define LDA   520   // 512 halfs + 8 halfs pad (16B-aligned rows, de-conflicted banks)

// Fast tanh: native v_tanh_f32 on gfx1250 if the builtin exists, else one v_exp_f32.
__device__ __forceinline__ float fast_tanh(float x) {
#if __has_builtin(__builtin_amdgcn_tanhf)
    return __builtin_amdgcn_tanhf(x);
#else
    return 1.0f - 2.0f / (__expf(2.0f * x) + 1.0f);   // exact at both saturation limits
#endif
}

// ---------------------------------------------------------------------------
// Kernel 0: convert W (HDIM x HDIM f32, row-major [o,h]) to f16 once per call.
// ---------------------------------------------------------------------------
__global__ __launch_bounds__(256) void wcvt_kernel(const float* __restrict__ W,
                                                   _Float16* __restrict__ Wh) {
    int i = blockIdx.x * 256 + threadIdx.x;          // 65536 float4s
    float4 v = ((const float4*)W)[i];
    v4h h = {(_Float16)v.x, (_Float16)v.y, (_Float16)v.z, (_Float16)v.w};
    ((v4h*)Wh)[i] = h;
}

// ---------------------------------------------------------------------------
// Kernel 1: fused GEMM(tanh(X*W^T + b)) . context  ->  scores[B*S]
// One block = 128 rows of flattened (b,s). 8 waves x 16 rows each.
// A fragments live in VGPRs across the whole N loop (compiler hoists them);
// inner loop is B global_load_b128 clauses + v_wmma_f32_16x16x32_f16.
// ---------------------------------------------------------------------------
__global__ __launch_bounds__(256) void score_kernel(const float* __restrict__ X,
                                                    const _Float16* __restrict__ Wh,
                                                    const float* __restrict__ bias,
                                                    const float* __restrict__ ctx,
                                                    float* __restrict__ scores) {
    __shared__ _Float16 sA[MTILE * LDA];             // 130 KB, f16 X tile

    const int t = threadIdx.x;
    const long rowbase = (long)blockIdx.x * MTILE;

    // ---- stage 128x512 f32 tile -> f16 LDS (each thread: 64 float4) ----
    const float4* Xg = (const float4*)(X + rowbase * HDIM);
    #pragma unroll 4
    for (int j = 0; j < 64; ++j) {
        int idx4 = t + j * 256;                      // 0..16383
        int row  = idx4 >> 7;                        // 128 float4 per row
        int k4   = idx4 & 127;
        float4 v = Xg[row * 128 + k4];
        v4h h = {(_Float16)v.x, (_Float16)v.y, (_Float16)v.z, (_Float16)v.w};
        *(v4h*)&sA[row * LDA + k4 * 4] = h;
    }
    __syncthreads();

    const int wave = t >> 5;
    const int lane = t & 31;
    const int lrow = lane & 15;                      // row-in-tile (A) / col (B,C)
    const int hi   = lane >> 4;                      // lane half select
    const _Float16* aBase = &sA[(wave * 16 + lrow) * LDA];
    const int kaOff = hi * 16;                       // A: lanes16-31 hold K 16-31

    float sacc[8];
    #pragma unroll
    for (int r = 0; r < 8; ++r) sacc[r] = 0.0f;

    for (int nc = 0; nc < HDIM / 16; ++nc) {         // 32 N-chunks of 16 cols
        const int n = nc * 16 + lrow;
        // B[k,nn] = W[n, k]; lanes16-31 hold K 16..31 of the chunk
        const _Float16* bBase = Wh + (size_t)n * HDIM + hi * 16;

        v8f acc = {};
        #pragma unroll
        for (int kc = 0; kc < HDIM / 32; ++kc) {     // 16 K-chunks of 32
            const int ka = kc * 32 + kaOff;
            union { v16h v; v8h h[2]; } ua;
            ua.h[0] = *(const v8h*)(aBase + ka);      // K ka..ka+7
            ua.h[1] = *(const v8h*)(aBase + ka + 8);  // K ka+8..ka+15
            v16h Bm = *(const v16h*)(bBase + kc * 32);// 32B contiguous
            acc = __builtin_amdgcn_wmma_f32_16x16x32_f16(
                false, ua.v, false, Bm, (short)0, acc, false, false);
        }

        // epilogue: tanh(acc + bias[n]) * ctx[n]; accumulate LANE-LOCAL partials
        // (reduction across the 16 N lanes is deferred to after the N loop)
        const float bn = bias[n];
        const float cn = ctx[n];
        #pragma unroll
        for (int r = 0; r < 8; ++r) {
            sacc[r] += fast_tanh(acc[r] + bn) * cn;
        }
    }

    // one butterfly reduction per row (32 shuffles/wave instead of 1024)
    #pragma unroll
    for (int r = 0; r < 8; ++r) {
        float v = sacc[r];
        v += __shfl_xor(v, 1, 16);
        v += __shfl_xor(v, 2, 16);
        v += __shfl_xor(v, 4, 16);
        v += __shfl_xor(v, 8, 16);
        sacc[r] = v;                                 // lanes 0 / 16 hold row r / r+8
    }

    const long outb = rowbase + wave * 16;
    if (lane == 0) {
        #pragma unroll
        for (int r = 0; r < 8; ++r) scores[outb + r] = sacc[r];
    } else if (lane == 16) {
        #pragma unroll
        for (int r = 0; r < 8; ++r) scores[outb + 8 + r] = sacc[r];
    }
}

// ---------------------------------------------------------------------------
// Kernel 2: masked softmax over S per batch row (in-place on scores -> probs)
// ---------------------------------------------------------------------------
__global__ __launch_bounds__(256) void softmax_kernel(const int* __restrict__ mask,
                                                      float* __restrict__ p) {
    const int b = blockIdx.x, t = threadIdx.x;
    const int base = b * SEQ;
    __shared__ float red[8];

    float vals[16];
    float m = -3.4e38f;
    #pragma unroll
    for (int i = 0; i < 16; ++i) {
        int s = t + i * 256;
        float sc = p[base + s];
        float v = (mask[base + s] == 0) ? -1e-32f : sc;   // faithful NEG_FILL
        vals[i] = v;
        m = fmaxf(m, v);
    }
    for (int off = 16; off > 0; off >>= 1) m = fmaxf(m, __shfl_xor(m, off, 32));
    if ((t & 31) == 0) red[t >> 5] = m;
    __syncthreads();
    if (t == 0) { float x = red[0]; for (int i = 1; i < 8; ++i) x = fmaxf(x, red[i]); red[0] = x; }
    __syncthreads();
    m = red[0];
    __syncthreads();

    float ssum = 0.0f;
    #pragma unroll
    for (int i = 0; i < 16; ++i) ssum += __expf(vals[i] - m);
    for (int off = 16; off > 0; off >>= 1) ssum += __shfl_xor(ssum, off, 32);
    if ((t & 31) == 0) red[t >> 5] = ssum;
    __syncthreads();
    if (t == 0) { float x = 0.f; for (int i = 0; i < 8; ++i) x += red[i]; red[0] = x; }
    __syncthreads();
    const float inv = 1.0f / red[0];

    #pragma unroll
    for (int i = 0; i < 16; ++i) {
        int s = t + i * 256;
        p[base + s] = __expf(vals[i] - m) * inv;
    }
}

// ---------------------------------------------------------------------------
// Kernel 3: partial pooling  part[b,c,h] = sum_{s in chunk c} p[b,s]*X[b,s,h]
// grid = B * 8 chunks (512 s each); 256 threads, 2 h per thread (float2).
// ---------------------------------------------------------------------------
__global__ __launch_bounds__(256) void pool_partial_kernel(const float* __restrict__ X,
                                                           const float* __restrict__ p,
                                                           float* __restrict__ part) {
    const int b = blockIdx.x >> 3;
    const int c = blockIdx.x & 7;
    const int t = threadIdx.x;
    __shared__ float sp[512];
    const int sbase = c * 512;
    sp[t]       = p[b * SEQ + sbase + t];
    sp[t + 256] = p[b * SEQ + sbase + t + 256];
    __syncthreads();

    const float2* Xg = (const float2*)(X + ((size_t)b * SEQ + sbase) * HDIM);
    float2 acc = {0.f, 0.f};
    for (int s = 0; s < 512; ++s) {
        float w  = sp[s];
        float2 x = Xg[(size_t)s * 256 + t];
        acc.x += w * x.x;
        acc.y += w * x.y;
    }
    ((float2*)part)[((size_t)(b * 8 + c)) * 256 + t] = acc;
}

// ---------------------------------------------------------------------------
// Kernel 4: final reduce over the 8 S-chunks -> out[B,H]
// ---------------------------------------------------------------------------
__global__ __launch_bounds__(256) void pool_reduce_kernel(const float* __restrict__ part,
                                                          float* __restrict__ out) {
    int i = blockIdx.x * 256 + threadIdx.x;          // 0..16383
    int b = i >> 9, h = i & 511;
    float a = 0.f;
    #pragma unroll
    for (int c = 0; c < 8; ++c) a += part[((size_t)(b * 8 + c)) * 512 + h];
    out[i] = a;
}

// ---------------------------------------------------------------------------
extern "C" void kernel_launch(void* const* d_in, const int* in_sizes, int n_in,
                              void* d_out, int out_size, void* d_ws, size_t ws_size,
                              hipStream_t stream) {
    const float* X    = (const float*)d_in[0];   // [B,S,H] f32
    const int*   mask = (const int*)d_in[1];     // [B,S] i32
    const float* W    = (const float*)d_in[2];   // [H,H] f32
    const float* bias = (const float*)d_in[3];   // [H]   f32
    const float* ctx  = (const float*)d_in[4];   // [H]   f32
    float* out = (float*)d_out;                  // [B,H] f32

    char* ws = (char*)d_ws;
    _Float16* Wh    = (_Float16*)ws;                    // 512 KB
    float*    sc    = (float*)(ws + (512u << 10));      // 512 KB scores/probs
    float*    part  = (float*)(ws + (1024u << 10));     // 512 KB partial pools

    wcvt_kernel<<<HDIM * HDIM / (256 * 4), 256, 0, stream>>>(W, Wh);
    score_kernel<<<BATCH * SEQ / MTILE, 256, 0, stream>>>(X, Wh, bias, ctx, sc);
    softmax_kernel<<<BATCH, 256, 0, stream>>>(mask, sc);
    pool_partial_kernel<<<BATCH * 8, 256, 0, stream>>>(X, sc, part);
    pool_reduce_kernel<<<BATCH * HDIM / 256, 256, 0, stream>>>(part, out);
}
